// GRAPHEDX_xor_on_edge_label_62388694942292
// MI455X (gfx1250) — matrix operand order repaired and checked by reading, not compiled, not run on hardware
//
#include <hip/hip_runtime.h>
#include <hip/hip_bf16.h>

// ---------------------------------------------------------------------------
// GraphEdX (xor-on-edge-label) forward pass for gfx1250 (MI455X).
// All MLP GEMMs go through v_wmma_f32_16x16x32_f16 (f16 A/B, f32 accum).
// ---------------------------------------------------------------------------

typedef __attribute__((ext_vector_type(16))) _Float16 v16h;
typedef __attribute__((ext_vector_type(8)))  float    v8f;

// ---- problem constants ----
static constexpr int NN  = 20;      // nodes per graph
static constexpr int NB  = 64;      // batch (query,corpus) pairs
static constexpr int NC2 = 190;     // N*(N-1)/2
static constexpr int TN  = 2560;    // total nodes (G*N)
static constexpr int NE  = 20480;   // edges
static constexpr int DD  = 32;      // node embedding dim
static constexpr int DM  = 64;      // message dim
static constexpr float SINK_TEMP = 0.05f;

// ---- workspace layout (floats) ----
static constexpr size_t OF_NF   = 0;        // 2560*32
static constexpr size_t OF_EF   = 81920;    // 20480*32
static constexpr size_t OF_AGG  = 737280;   // 2560*64
static constexpr size_t OF_RAGG = 901120;   // 2560*64
static constexpr size_t OF_T    = 1064960;  // 2560*20
static constexpr size_t OF_PLAN = 1116160;  // 64*20*20
static constexpr size_t OF_QE   = 1141760;  // 64*190*64
static constexpr size_t OF_CE   = 1920000;  // 64*190*64
static constexpr size_t OF_FEND = 2698240;
// ---- workspace layout (halfs, after float region): transposed padded weights ----
static constexpr size_t H_ENCN = 0,     H_ENCE = 1024,  H_MSG1 = 2048,  H_MSG2 = 8192;
static constexpr size_t H_RMSG1= 12288, H_RMSG2= 18432, H_UPD1 = 22528, H_UPD2 = 32768;
static constexpr size_t H_SINK1= 34816, H_SINK2= 35840, H_LRL1 = 36864, H_LRL2 = 43008;
static constexpr size_t H_HEND = 47104;

// ---------------------------------------------------------------------------
// WMMA fragment helpers (CDNA5 16x16x32 f16 layouts, wave32)
// ---------------------------------------------------------------------------
__device__ __forceinline__ int a_kbase(int lane, int i) {
    // 16-bit A-matrix 16x32: VGPR i, lane -> K index
    return ((i >> 2) << 4) + ((lane >> 4) << 3) + ((i & 3) << 1);
}
__device__ __forceinline__ int b_kbase(int lane, int i) {
    // 16-bit B-matrix 32x16 (stored as Wt[n][k]): same K order as A (lanes 0-15 K=0..7,16..23; 16-31 K=8..15,24..31)
    return ((i >> 2) << 4) + ((lane >> 4) << 3) + ((i & 3) << 1);
}

__device__ __forceinline__ v16h frag_a_f32(const float* rowp, int kvalid) {
    const int lane = threadIdx.x & 31;
    v16h a;
#pragma unroll
    for (int i = 0; i < 8; ++i) {
        int k = a_kbase(lane, i);
        a[2*i]   = (_Float16)((k     < kvalid) ? rowp[k]     : 0.f);
        a[2*i+1] = (_Float16)((k + 1 < kvalid) ? rowp[k + 1] : 0.f);
    }
    return a;
}
__device__ __forceinline__ v16h frag_a_f16(const _Float16* rowp) {
    const int lane = threadIdx.x & 31;
    v16h a;
#pragma unroll
    for (int i = 0; i < 8; ++i) {
        int k = a_kbase(lane, i);
        a[2*i] = rowp[k]; a[2*i+1] = rowp[k + 1];
    }
    return a;
}
__device__ __forceinline__ v16h frag_b_f16(const _Float16* rowp) {
    const int lane = threadIdx.x & 31;
    v16h b;
#pragma unroll
    for (int i = 0; i < 8; ++i) {
        int k = b_kbase(lane, i);
        b[2*i] = rowp[k]; b[2*i+1] = rowp[k + 1];
    }
    return b;
}
__device__ __forceinline__ v16h frag_a_scalar_k0(float v) {
    // A fragment whose only nonzero element is K==0 of this lane's row.
    const int lane = threadIdx.x & 31;
    v16h a = {};
    if (lane < 16) a[0] = (_Float16)v;   // lanes 0-15, VGPR0 low half = K 0
    return a;
}
__device__ __forceinline__ v8f wmma16(v16h a, v16h b, v8f c) {
    return __builtin_amdgcn_wmma_f32_16x16x32_f16(false, a, false, b, (short)0, c,
                                                  false, false);
}

// ---------------------------------------------------------------------------
// setup kernels
// ---------------------------------------------------------------------------
__global__ void k_pairs(int* __restrict__ srcp, int* __restrict__ dstp) {
    int p = blockIdx.x * blockDim.x + threadIdx.x;
    if (p >= NC2) return;
    int i = 0, rem = p;
    while (rem >= (NN - 1 - i)) { rem -= (NN - 1 - i); ++i; }
    srcp[p] = i; dstp[p] = i + 1 + rem;
}

// W (K x Nn, f32 row-major) -> Wt (Npad x Kpad, f16, Wt[n][k] = W[k][n], zero padded)
__global__ void k_pack_wt(const float* __restrict__ W, _Float16* __restrict__ Wt,
                          int K, int Nn, int Kpad, int Npad) {
    int idx = blockIdx.x * blockDim.x + threadIdx.x;
    if (idx >= Kpad * Npad) return;
    int n = idx / Kpad, k = idx % Kpad;
    float v = (k < K && n < Nn) ? W[(size_t)k * Nn + n] : 0.f;
    Wt[idx] = (_Float16)v;
}

__global__ void k_zero(float* __restrict__ p, int n) {
    int i = blockIdx.x * blockDim.x + threadIdx.x;
    if (i < n) p[i] = 0.f;
}

// ---------------------------------------------------------------------------
// encoders: Y[rows x 32] = X[rows x kvalid] @ W + b     (one wave / 16 rows)
// ---------------------------------------------------------------------------
__global__ __launch_bounds__(32) void k_encode(
    const float* __restrict__ X, int ldx, int kvalid,
    const _Float16* __restrict__ Wt, const float* __restrict__ bias,
    float* __restrict__ Y) {
    const int lane = threadIdx.x, r = lane & 15, half = lane >> 4;
    const int m0 = blockIdx.x * 16;
    v16h a = frag_a_f32(X + (size_t)(m0 + r) * ldx, kvalid);
#pragma unroll
    for (int nt = 0; nt < 2; ++nt) {
        v8f c = {};
        c = wmma16(a, frag_b_f16(Wt + (size_t)(nt * 16 + r) * 32), c);
        int col = nt * 16 + r;
        float bb = bias[col];
#pragma unroll
        for (int vr = 0; vr < 8; ++vr)
            Y[(size_t)(m0 + vr + 8 * half) * DD + col] = c[vr] + bb;
    }
}

// ---------------------------------------------------------------------------
// message step: per 16-edge tile, msg+rmsg MLPs (96->64->64) + atomic scatter
// ---------------------------------------------------------------------------
__global__ __launch_bounds__(32) void k_msg(
    const float* __restrict__ nf, const float* __restrict__ ef,
    const int* __restrict__ from_idx, const int* __restrict__ to_idx,
    const _Float16* __restrict__ w1t, const float* __restrict__ b1,
    const _Float16* __restrict__ w2t, const float* __restrict__ b2,
    const _Float16* __restrict__ rw1t, const float* __restrict__ rb1,
    const _Float16* __restrict__ rw2t, const float* __restrict__ rb2,
    float* __restrict__ agg, float* __restrict__ ragg) {
    __shared__ int sF[16], sT[16];
    __shared__ _Float16 hbuf[16][64];
    const int lane = threadIdx.x, r = lane & 15, half = lane >> 4;
    const int e0 = blockIdx.x * 16;
    if (lane < 16) { sF[lane] = from_idx[e0 + lane]; sT[lane] = to_idx[e0 + lane]; }
    __syncthreads();
    const int fi = sF[r], ti = sT[r];
    v16h af = frag_a_f32(nf + (size_t)fi * DD, DD);
    v16h at = frag_a_f32(nf + (size_t)ti * DD, DD);
    v16h ae = frag_a_f32(ef + (size_t)(e0 + r) * DD, DD);

    for (int dir = 0; dir < 2; ++dir) {
        const _Float16* W1 = dir ? rw1t : w1t;
        const float*    B1 = dir ? rb1  : b1;
        const _Float16* W2 = dir ? rw2t : w2t;
        const float*    B2 = dir ? rb2  : b2;
        float* outp   = dir ? ragg : agg;
        const int* seg = dir ? sF : sT;
        v16h x0 = dir ? at : af;
        v16h x1 = dir ? af : at;

        v8f acc[4];
#pragma unroll
        for (int nt = 0; nt < 4; ++nt) {
            acc[nt] = {};
            const _Float16* wr = W1 + (size_t)(nt * 16 + r) * 96;
            acc[nt] = wmma16(x0, frag_b_f16(wr +  0), acc[nt]);
            acc[nt] = wmma16(x1, frag_b_f16(wr + 32), acc[nt]);
            acc[nt] = wmma16(ae, frag_b_f16(wr + 64), acc[nt]);
        }
        __syncthreads();
#pragma unroll
        for (int nt = 0; nt < 4; ++nt) {
            int col = nt * 16 + r;
            float bb = B1[col];
#pragma unroll
            for (int vr = 0; vr < 8; ++vr) {
                float v = acc[nt][vr] + bb;
                hbuf[vr + 8 * half][col] = (_Float16)(v > 0.f ? v : 0.f);
            }
        }
        __syncthreads();
        v16h h0 = frag_a_f16(&hbuf[r][0]);
        v16h h1 = frag_a_f16(&hbuf[r][32]);
#pragma unroll
        for (int nt = 0; nt < 4; ++nt) {
            v8f o = {};
            const _Float16* wr = W2 + (size_t)(nt * 16 + r) * 64;
            o = wmma16(h0, frag_b_f16(wr +  0), o);
            o = wmma16(h1, frag_b_f16(wr + 32), o);
            int col = nt * 16 + r;
            float bb = B2[col];
#pragma unroll
            for (int vr = 0; vr < 8; ++vr) {
                int row = vr + 8 * half;
                __hip_atomic_fetch_add(&outp[(size_t)seg[row] * DM + col],
                                       o[vr] + bb,
                                       __ATOMIC_RELAXED, __HIP_MEMORY_SCOPE_AGENT);
            }
        }
        __syncthreads();
    }
}

// ---------------------------------------------------------------------------
// node update: nf += mlp2([agg|ragg|nf] 160 -> 64 -> 32)
// ---------------------------------------------------------------------------
__global__ __launch_bounds__(32) void k_update(
    float* __restrict__ nf, const float* __restrict__ agg, const float* __restrict__ ragg,
    const _Float16* __restrict__ w1t, const float* __restrict__ b1,
    const _Float16* __restrict__ w2t, const float* __restrict__ b2) {
    __shared__ _Float16 hbuf[16][64];
    const int lane = threadIdx.x, r = lane & 15, half = lane >> 4;
    const int m0 = blockIdx.x * 16;
    const float* ap = agg  + (size_t)(m0 + r) * DM;
    const float* rp = ragg + (size_t)(m0 + r) * DM;
    const float* np = nf   + (size_t)(m0 + r) * DD;
    v16h A0 = frag_a_f32(ap, 32),  A1 = frag_a_f32(ap + 32, 32);
    v16h A2 = frag_a_f32(rp, 32),  A3 = frag_a_f32(rp + 32, 32);
    v16h A4 = frag_a_f32(np, 32);
    v8f acc[4];
#pragma unroll
    for (int nt = 0; nt < 4; ++nt) {
        acc[nt] = {};
        const _Float16* wr = w1t + (size_t)(nt * 16 + r) * 160;
        acc[nt] = wmma16(A0, frag_b_f16(wr +   0), acc[nt]);
        acc[nt] = wmma16(A1, frag_b_f16(wr +  32), acc[nt]);
        acc[nt] = wmma16(A2, frag_b_f16(wr +  64), acc[nt]);
        acc[nt] = wmma16(A3, frag_b_f16(wr +  96), acc[nt]);
        acc[nt] = wmma16(A4, frag_b_f16(wr + 128), acc[nt]);
    }
#pragma unroll
    for (int nt = 0; nt < 4; ++nt) {
        int col = nt * 16 + r;
        float bb = b1[col];
#pragma unroll
        for (int vr = 0; vr < 8; ++vr) {
            float v = acc[nt][vr] + bb;
            hbuf[vr + 8 * half][col] = (_Float16)(v > 0.f ? v : 0.f);
        }
    }
    __syncthreads();
    v16h h0 = frag_a_f16(&hbuf[r][0]);
    v16h h1 = frag_a_f16(&hbuf[r][32]);
#pragma unroll
    for (int nt = 0; nt < 2; ++nt) {
        v8f o = {};
        const _Float16* wr = w2t + (size_t)(nt * 16 + r) * 64;
        o = wmma16(h0, frag_b_f16(wr +  0), o);
        o = wmma16(h1, frag_b_f16(wr + 32), o);
        int col = nt * 16 + r;
        float bb = b2[col];
#pragma unroll
        for (int vr = 0; vr < 8; ++vr)
            nf[(size_t)(m0 + vr + 8 * half) * DD + col] += o[vr] + bb;
    }
}

// ---------------------------------------------------------------------------
// sinkhorn MLP: t[2560 x 20] = mlp2(nf, 32 -> 20 -> 20)  (N padded to 32)
// ---------------------------------------------------------------------------
__global__ __launch_bounds__(32) void k_sinkmlp(
    const float* __restrict__ nf,
    const _Float16* __restrict__ w1t, const float* __restrict__ b1,
    const _Float16* __restrict__ w2t, const float* __restrict__ b2,
    float* __restrict__ t) {
    __shared__ _Float16 hbuf[16][32];
    const int lane = threadIdx.x, r = lane & 15, half = lane >> 4;
    const int m0 = blockIdx.x * 16;
    v16h a = frag_a_f32(nf + (size_t)(m0 + r) * DD, 32);
#pragma unroll
    for (int nt = 0; nt < 2; ++nt) {
        v8f c = {};
        c = wmma16(a, frag_b_f16(w1t + (size_t)(nt * 16 + r) * 32), c);
        int col = nt * 16 + r;
        float bb = (col < NN) ? b1[col] : 0.f;
#pragma unroll
        for (int vr = 0; vr < 8; ++vr) {
            float v = c[vr] + bb;
            v = (col < NN && v > 0.f) ? v : 0.f;
            hbuf[vr + 8 * half][col] = (_Float16)v;
        }
    }
    __syncthreads();
    v16h h = frag_a_f16(&hbuf[r][0]);
#pragma unroll
    for (int nt = 0; nt < 2; ++nt) {
        v8f o = {};
        o = wmma16(h, frag_b_f16(w2t + (size_t)(nt * 16 + r) * 32), o);
        int col = nt * 16 + r;
        if (col < NN) {
            float bb = b2[col];
#pragma unroll
            for (int vr = 0; vr < 8; ++vr)
                t[(size_t)(m0 + vr + 8 * half) * NN + col] = o[vr] + bb;
        }
    }
}

// ---------------------------------------------------------------------------
// sinkhorn: plan[b] = sinkhorn(-cdist_l1(tq,tc)/TEMP), 20 iters, 20x20 block
// ---------------------------------------------------------------------------
__global__ void k_sinkhorn(const float* __restrict__ t, float* __restrict__ plan) {
    __shared__ float la[NN][NN];
    __shared__ float red[NN];
    const int b = blockIdx.x, j = threadIdx.x, i = threadIdx.y;
    const float* qi = t + ((size_t)(2 * b)     * NN + i) * NN;
    const float* cj = t + ((size_t)(2 * b + 1) * NN + j) * NN;
    float s = 0.f;
    for (int k = 0; k < NN; ++k) s += fabsf(qi[k] - cj[k]);
    la[i][j] = -s * (1.0f / SINK_TEMP);
    __syncthreads();
    for (int it = 0; it < 20; ++it) {
        if (j == 0) {                      // logsumexp over columns (axis=2)
            float m = -3.0e38f;
            for (int jj = 0; jj < NN; ++jj) m = fmaxf(m, la[i][jj]);
            float ss = 0.f;
            for (int jj = 0; jj < NN; ++jj) ss += expf(la[i][jj] - m);
            red[i] = m + logf(ss);
        }
        __syncthreads();
        la[i][j] -= red[i];
        __syncthreads();
        if (i == 0) {                      // logsumexp over rows (axis=1)
            float m = -3.0e38f;
            for (int ii = 0; ii < NN; ++ii) m = fmaxf(m, la[ii][j]);
            float ss = 0.f;
            for (int ii = 0; ii < NN; ++ii) ss += expf(la[ii][j] - m);
            red[j] = m + logf(ss);
        }
        __syncthreads();
        la[i][j] -= red[j];
        __syncthreads();
    }
    plan[((size_t)b * NN + i) * NN + j] = expf(la[i][j]);
}

// ---------------------------------------------------------------------------
// nc2 pair embeddings: out = f + b,  f/b = mlp2([s|d|e] or [d|s|e], 65->64->64)
// ---------------------------------------------------------------------------
__global__ __launch_bounds__(32) void k_nc2(
    const float* __restrict__ nf, const float* __restrict__ adj, int side,
    const int* __restrict__ srcp, const int* __restrict__ dstp,
    const _Float16* __restrict__ w1t, const float* __restrict__ b1,
    const _Float16* __restrict__ w2t, const float* __restrict__ b2,
    float* __restrict__ outp) {
    __shared__ _Float16 hbuf[16][64];
    const int lane = threadIdx.x, r = lane & 15, half = lane >> 4;
    const int pr0 = blockIdx.x * 16, pr = pr0 + r;
    const int b = pr / NC2, p = pr % NC2;
    const int si = srcp[p], di = dstp[p];
    const int g = 2 * b + side;
    v16h as = frag_a_f32(nf + ((size_t)g * NN + si) * DD, 32);
    v16h ad = frag_a_f32(nf + ((size_t)g * NN + di) * DD, 32);
    v16h ae = frag_a_scalar_k0(adj[((size_t)b * NN + si) * NN + di]);
    v8f res[4];
#pragma unroll
    for (int nt = 0; nt < 4; ++nt) res[nt] = {};
    for (int pass = 0; pass < 2; ++pass) {
        v16h x0 = pass ? ad : as;
        v16h x1 = pass ? as : ad;
        v8f acc[4];
#pragma unroll
        for (int nt = 0; nt < 4; ++nt) {
            acc[nt] = {};
            const _Float16* wr = w1t + (size_t)(nt * 16 + r) * 96;
            acc[nt] = wmma16(x0, frag_b_f16(wr +  0), acc[nt]);
            acc[nt] = wmma16(x1, frag_b_f16(wr + 32), acc[nt]);
            acc[nt] = wmma16(ae, frag_b_f16(wr + 64), acc[nt]);
        }
        __syncthreads();
#pragma unroll
        for (int nt = 0; nt < 4; ++nt) {
            int col = nt * 16 + r;
            float bb = b1[col];
#pragma unroll
            for (int vr = 0; vr < 8; ++vr) {
                float v = acc[nt][vr] + bb;
                hbuf[vr + 8 * half][col] = (_Float16)(v > 0.f ? v : 0.f);
            }
        }
        __syncthreads();
        v16h h0 = frag_a_f16(&hbuf[r][0]);
        v16h h1 = frag_a_f16(&hbuf[r][32]);
#pragma unroll
        for (int nt = 0; nt < 4; ++nt) {
            const _Float16* wr = w2t + (size_t)(nt * 16 + r) * 64;
            res[nt] = wmma16(h0, frag_b_f16(wr +  0), res[nt]);
            res[nt] = wmma16(h1, frag_b_f16(wr + 32), res[nt]);
        }
        __syncthreads();
    }
#pragma unroll
    for (int nt = 0; nt < 4; ++nt) {
        int col = nt * 16 + r;
        float bb = 2.f * b2[col];            // f + b each add b2 once
#pragma unroll
        for (int vr = 0; vr < 8; ++vr)
            outp[(size_t)(pr0 + vr + 8 * half) * DM + col] = res[nt][vr] + bb;
    }
}

// ---------------------------------------------------------------------------
// final: edge_align + node_align + 0.5*label_align per batch pair
// ---------------------------------------------------------------------------
__global__ __launch_bounds__(256) void k_final(
    const float* __restrict__ nf, const float* __restrict__ ohl,
    const float* __restrict__ qadj, const float* __restrict__ cadj,
    const float* __restrict__ plan, const float* __restrict__ qe,
    const float* __restrict__ ce, const int* __restrict__ srcp,
    const int* __restrict__ dstp, const int* __restrict__ gsz,
    float* __restrict__ out) {
    __shared__ float shP[NN][NN];
    __shared__ int ssrc[NC2], sdst[NC2], saq[NC2], sac[NC2];
    __shared__ float red[256];
    const int b = blockIdx.x, tid = threadIdx.x;
    for (int idx = tid; idx < NN * NN; idx += 256)
        shP[idx / NN][idx % NN] = plan[(size_t)b * NN * NN + idx];
    for (int idx = tid; idx < NC2; idx += 256) {
        int s = srcp[idx], d = dstp[idx];
        ssrc[idx] = s; sdst[idx] = d;
        saq[idx] = (int)qadj[((size_t)b * NN + s) * NN + d];
        sac[idx] = (int)cadj[((size_t)b * NN + s) * NN + d];
    }
    __syncthreads();
    const float* qeB = qe + (size_t)b * NC2 * DM;
    const float* ceB = ce + (size_t)b * NC2 * DM;
    float acc = 0.f;
    for (int idx = tid; idx < NC2 * NC2; idx += 256) {
        int p = idx / NC2, q = idx % NC2;
        if (saq[p] != sac[q]) {                                 // xor mask
            int sp = ssrc[p], dp = sdst[p], sq = ssrc[q], dq = sdst[q];
            float ew = shP[sp][sq] * shP[dp][dq] + shP[sp][dq] * shP[dp][sq];
            const float* a = qeB + (size_t)p * DM;
            const float* c = ceB + (size_t)q * DM;
            float dist = 0.f;
            for (int k = 0; k < DM; ++k) dist += fabsf(a[k] - c[k]);
            acc += dist * ew;
        }
    }
    const float* qn = nf  + (size_t)(2 * b)     * NN * DD;
    const float* cn = nf  + (size_t)(2 * b + 1) * NN * DD;
    const float* ql = ohl + (size_t)(2 * b)     * NN * 16;
    const float* cl = ohl + (size_t)(2 * b + 1) * NN * 16;
    const int qs = gsz[2 * b], cs = gsz[2 * b + 1];
    for (int idx = tid; idx < NN * NN; idx += 256) {
        int i = idx / NN, j = idx % NN;
        float d32 = 0.f;
        for (int k = 0; k < DD; ++k) d32 += fabsf(qn[i * DD + k] - cn[j * DD + k]);
        float d16 = 0.f;
        for (int k = 0; k < 16; ++k) d16 += fabsf(ql[i * 16 + k] - cl[j * 16 + k]);
        float pij = shP[i][j];
        float ind = (i < qs && j < cs) ? 1.f : 0.f;
        acc += pij * d32 + 0.5f * pij * d16 * ind;
    }
    red[tid] = acc;
    __syncthreads();
    for (int s = 128; s > 0; s >>= 1) {
        if (tid < s) red[tid] += red[tid + s];
        __syncthreads();
    }
    if (tid == 0) out[b] = red[0];
}

// ---------------------------------------------------------------------------
// host driver
// ---------------------------------------------------------------------------
extern "C" void kernel_launch(void* const* d_in, const int* in_sizes, int n_in,
                              void* d_out, int out_size, void* d_ws, size_t ws_size,
                              hipStream_t stream) {
    (void)in_sizes; (void)n_in; (void)out_size; (void)ws_size;
    // setup_inputs() dict order (params nested in declaration order)
    const float* node_features = (const float*)d_in[0];
    const float* edge_features = (const float*)d_in[1];
    const float* one_hot       = (const float*)d_in[2];
    const float* qadj          = (const float*)d_in[3];
    const float* cadj          = (const float*)d_in[4];
    const float* enc_nw = (const float*)d_in[5];  const float* enc_nb = (const float*)d_in[6];
    const float* enc_ew = (const float*)d_in[7];  const float* enc_eb = (const float*)d_in[8];
    const float* msg_w1 = (const float*)d_in[9];  const float* msg_b1 = (const float*)d_in[10];
    const float* msg_w2 = (const float*)d_in[11]; const float* msg_b2 = (const float*)d_in[12];
    const float* rmsg_w1= (const float*)d_in[13]; const float* rmsg_b1= (const float*)d_in[14];
    const float* rmsg_w2= (const float*)d_in[15]; const float* rmsg_b2= (const float*)d_in[16];
    const float* upd_w1 = (const float*)d_in[17]; const float* upd_b1 = (const float*)d_in[18];
    const float* upd_w2 = (const float*)d_in[19]; const float* upd_b2 = (const float*)d_in[20];
    const float* sink_w1= (const float*)d_in[21]; const float* sink_b1= (const float*)d_in[22];
    const float* sink_w2= (const float*)d_in[23]; const float* sink_b2= (const float*)d_in[24];
    const float* lrl_w1 = (const float*)d_in[25]; const float* lrl_b1 = (const float*)d_in[26];
    const float* lrl_w2 = (const float*)d_in[27]; const float* lrl_b2 = (const float*)d_in[28];
    const int* from_idx = (const int*)d_in[29];
    const int* to_idx   = (const int*)d_in[30];
    /* d_in[31] = graph_idx (implied by layout) */
    const int* graph_sizes = (const int*)d_in[32];
    float* out = (float*)d_out;

    float* wsf = (float*)d_ws;
    _Float16* wsh = (_Float16*)(wsf + OF_FEND);
    int* wsi = (int*)(wsh + H_HEND);
    int* SRCp = wsi; int* DSTp = wsi + NC2;

    float* nf   = wsf + OF_NF;   float* ef   = wsf + OF_EF;
    float* agg  = wsf + OF_AGG;  float* ragg = wsf + OF_RAGG;
    float* t    = wsf + OF_T;    float* plan = wsf + OF_PLAN;
    float* qe   = wsf + OF_QE;   float* ce   = wsf + OF_CE;

    k_pairs<<<1, 256, 0, stream>>>(SRCp, DSTp);

    struct PW { const float* w; size_t off; int K, Nn, Kp, Np; };
    const PW pws[12] = {
        {enc_nw,  H_ENCN,  16, 32,  32, 32}, {enc_ew,  H_ENCE,   8, 32,  32, 32},
        {msg_w1,  H_MSG1,  96, 64,  96, 64}, {msg_w2,  H_MSG2,  64, 64,  64, 64},
        {rmsg_w1, H_RMSG1, 96, 64,  96, 64}, {rmsg_w2, H_RMSG2, 64, 64,  64, 64},
        {upd_w1,  H_UPD1, 160, 64, 160, 64}, {upd_w2,  H_UPD2,  64, 32,  64, 32},
        {sink_w1, H_SINK1, 32, 20,  32, 32}, {sink_w2, H_SINK2, 20, 20,  32, 32},
        {lrl_w1,  H_LRL1,  65, 64,  96, 64}, {lrl_w2,  H_LRL2,  64, 64,  64, 64},
    };
    for (int i = 0; i < 12; ++i) {
        int tot = pws[i].Kp * pws[i].Np;
        k_pack_wt<<<(tot + 255) / 256, 256, 0, stream>>>(
            pws[i].w, wsh + pws[i].off, pws[i].K, pws[i].Nn, pws[i].Kp, pws[i].Np);
    }

    k_encode<<<TN / 16, 32, 0, stream>>>(node_features, 16, 16, wsh + H_ENCN, enc_nb, nf);
    k_encode<<<NE / 16, 32, 0, stream>>>(edge_features,  8,  8, wsh + H_ENCE, enc_eb, ef);

    for (int step = 0; step < 5; ++step) {
        k_zero<<<(2 * TN * DM + 255) / 256, 256, 0, stream>>>(agg, 2 * TN * DM);
        k_msg<<<NE / 16, 32, 0, stream>>>(nf, ef, from_idx, to_idx,
            wsh + H_MSG1, msg_b1, wsh + H_MSG2, msg_b2,
            wsh + H_RMSG1, rmsg_b1, wsh + H_RMSG2, rmsg_b2, agg, ragg);
        k_update<<<TN / 16, 32, 0, stream>>>(nf, agg, ragg,
            wsh + H_UPD1, upd_b1, wsh + H_UPD2, upd_b2);
    }

    k_sinkmlp<<<TN / 16, 32, 0, stream>>>(nf, wsh + H_SINK1, sink_b1,
                                          wsh + H_SINK2, sink_b2, t);
    k_sinkhorn<<<NB, dim3(NN, NN), 0, stream>>>(t, plan);

    k_nc2<<<(NB * NC2) / 16, 32, 0, stream>>>(nf, qadj, 0, SRCp, DSTp,
        wsh + H_LRL1, lrl_b1, wsh + H_LRL2, lrl_b2, qe);
    k_nc2<<<(NB * NC2) / 16, 32, 0, stream>>>(nf, cadj, 1, SRCp, DSTp,
        wsh + H_LRL1, lrl_b1, wsh + H_LRL2, lrl_b2, ce);

    k_final<<<NB, 256, 0, stream>>>(nf, one_hot, qadj, cadj, plan, qe, ce,
                                    SRCp, DSTp, graph_sizes, out);
}
